// LDS_Late_35605278884572
// MI455X (gfx1250) — compile-verified, matrix operand order checked by reading, not run on hardware
//
#include <hip/hip_runtime.h>
#include <stdint.h>
#include <stddef.h>

// ---------------------------------------------------------------------------
// LDS_Late forward for MI455X (gfx1250, wave32, WMMA).
//   K0  : convert X -> bf16; transpose B,C,M -> N-major bf16
//   K1  : uB = X @ B        (32x32/wave register-blocked bf16 WMMA, f32 accum)
//   K2a : per-chunk local scan suffix sums      (chunk = 128, 16 chunks)
//   K2b : carry scan across chunks with A^128 (+h0)
//   K2c : expand chunks -> all_h (bf16)
//   K3  : out = sum_i M_i @ X(shift i)  +  (H @ C) shifted fwd by kx=10
//         (32x32/wave: 448 WMMAs per wave, 2x fragment reuse)
// ---------------------------------------------------------------------------

#define BSZ 16
#define SEQ 2048
#define IN_DIM 256
#define OUT_DIM 256
#define S_DIM 1024
#define KX 10
#define NCHUNK 16
#define CLEN 128          // NCHUNK * CLEN == SEQ
#define ROWS (BSZ * SEQ)  // 32768

typedef __bf16 bf16_t;
typedef bf16_t bf16x16 __attribute__((ext_vector_type(16)));
typedef float  f32x8   __attribute__((ext_vector_type(8)));

union Frag16 { bf16x16 v; uint32_t u[8]; };

__device__ __forceinline__ uint16_t f2bf(float f) {
  uint32_t u = __builtin_bit_cast(uint32_t, f);
  uint32_t r = u + 0x7FFFu + ((u >> 16) & 1u);   // round to nearest even
  return (uint16_t)(r >> 16);
}
__device__ __forceinline__ float bf2f(uint16_t h) {
  uint32_t u = ((uint32_t)h) << 16;
  return __builtin_bit_cast(float, u);
}

// K-index map for the 16-bit 16x32 A/B fragment layout (ISA 7.12.2):
// element pair v (0..7), half = lane>>4:
//   v<4 : K = 2v        + 8*half
//   v>=4: K = 16+2(v-4) + 8*half
// => per lane the 8 dwords form two contiguous 16B runs (backend emits b128).
__device__ __forceinline__ int kmap(int v, int half) {
  return ((v < 4) ? (2 * v) : (16 + 2 * (v - 4))) + 8 * half;
}

__device__ __forceinline__ void load_frag(Frag16& f, const uint16_t* row,
                                          int kbase, int half, bool ok) {
#pragma unroll
  for (int v = 0; v < 8; ++v) {
    int kk = kbase + kmap(v, half);
    f.u[v] = ok ? *(const uint32_t*)(row + kk) : 0u;
  }
}

// ------------------------------- K0: converts -------------------------------
__global__ void k0_cvtX(const float* __restrict__ X, uint16_t* __restrict__ Xbf, int n) {
  int i = blockIdx.x * blockDim.x + threadIdx.x;
  if (i < n) Xbf[i] = f2bf(X[i]);
}
// B: [IN_DIM][S_DIM] -> Bt[n=S_DIM][k=IN_DIM]
__global__ void k0_transB(const float* __restrict__ B, uint16_t* __restrict__ Bt) {
  int i = blockIdx.x * blockDim.x + threadIdx.x;       // i = n*256 + k
  if (i >= S_DIM * IN_DIM) return;
  int n = i >> 8, k = i & 255;
  Bt[i] = f2bf(B[(size_t)k * S_DIM + n]);
}
// C: [S_DIM][OUT_DIM] -> Ct[n=OUT_DIM][k=S_DIM]
__global__ void k0_transC(const float* __restrict__ C, uint16_t* __restrict__ Ct) {
  int i = blockIdx.x * blockDim.x + threadIdx.x;       // i = n*1024 + k
  if (i >= OUT_DIM * S_DIM) return;
  int n = i >> 10, k = i & 1023;
  Ct[i] = f2bf(C[(size_t)k * OUT_DIM + n]);
}
// M: [O][J][KX] -> Mt[i][o][j]
__global__ void k0_transM(const float* __restrict__ M, uint16_t* __restrict__ Mt) {
  int idx = blockIdx.x * blockDim.x + threadIdx.x;
  if (idx >= KX * OUT_DIM * IN_DIM) return;
  int i = idx / (OUT_DIM * IN_DIM);
  int rem = idx - i * (OUT_DIM * IN_DIM);
  int o = rem >> 8, j = rem & 255;
  Mt[idx] = f2bf(M[((size_t)o * IN_DIM + j) * KX + i]);
}

// ---------------- K1: uB = X @ B (32x32 per wave, WMMA) ---------------------
__global__ void k1_gemm_uB(const uint16_t* __restrict__ Xbf,
                           const uint16_t* __restrict__ Bt,
                           uint16_t* __restrict__ uBbf) {
  int wave = (int)((blockIdx.x * blockDim.x + threadIdx.x) >> 5);
  int lane = threadIdx.x & 31;
  int tileN = wave & 31;            // 32 col tiles of 32 (S_DIM/32)
  int tileM = wave >> 5;            // 1024 row tiles of 32 (ROWS/32)
  if (tileM >= ROWS / 32) return;
  int r = lane & 15, half = lane >> 4;
  int m0 = tileM << 5, n0 = tileN << 5;

  const uint16_t* ar0 = Xbf + (size_t)(m0 + r) * IN_DIM;
  const uint16_t* ar1 = ar0 + (size_t)16 * IN_DIM;
  const uint16_t* br0 = Bt  + (size_t)(n0 + r) * IN_DIM;
  const uint16_t* br1 = br0 + (size_t)16 * IN_DIM;

  f32x8 acc[2][2] = {};
#pragma unroll
  for (int kb = 0; kb < 8; ++kb) {
    Frag16 a0, a1, w0, w1;
    load_frag(a0, ar0, kb * 32, half, true);
    load_frag(a1, ar1, kb * 32, half, true);
    load_frag(w0, br0, kb * 32, half, true);
    load_frag(w1, br1, kb * 32, half, true);
    acc[0][0] = __builtin_amdgcn_wmma_f32_16x16x32_bf16(false, a0.v, false, w0.v, (short)0, acc[0][0], false, false);
    acc[0][1] = __builtin_amdgcn_wmma_f32_16x16x32_bf16(false, a0.v, false, w1.v, (short)0, acc[0][1], false, false);
    acc[1][0] = __builtin_amdgcn_wmma_f32_16x16x32_bf16(false, a1.v, false, w0.v, (short)0, acc[1][0], false, false);
    acc[1][1] = __builtin_amdgcn_wmma_f32_16x16x32_bf16(false, a1.v, false, w1.v, (short)0, acc[1][1], false, false);
  }
#pragma unroll
  for (int mi = 0; mi < 2; ++mi)
#pragma unroll
    for (int ni = 0; ni < 2; ++ni)
#pragma unroll
      for (int i = 0; i < 8; ++i)
        uBbf[(size_t)(m0 + mi * 16 + i + 8 * half) * S_DIM + n0 + ni * 16 + r]
            = f2bf(acc[mi][ni][i]);
}

// -------------------- K2: chunked diagonal affine scan ----------------------
// thread per (b, chunk, s): idx = ((b*NCHUNK)+c)*S_DIM + s
__global__ void k2a_local(const uint16_t* __restrict__ uBbf,
                          const float* __restrict__ A,
                          float* __restrict__ localEnd) {
  int idx = blockIdx.x * blockDim.x + threadIdx.x;
  if (idx >= BSZ * NCHUNK * S_DIM) return;
  int s = idx & (S_DIM - 1);
  int c = (idx >> 10) & (NCHUNK - 1);
  int b = idx >> 14;
  float a = A[s];
  float h = 0.0f;
  const uint16_t* p = uBbf + ((size_t)b * SEQ + (size_t)c * CLEN) * S_DIM + s;
#pragma unroll 4
  for (int k = 0; k < CLEN; ++k)
    h = fmaf(a, h, bf2f(p[(size_t)k * S_DIM]));
  localEnd[idx] = h;
}

__global__ void k2b_scan(const float* __restrict__ localEnd,
                         const float* __restrict__ A,
                         const float* __restrict__ h0,
                         float* __restrict__ carries) {
  int idx = blockIdx.x * blockDim.x + threadIdx.x;
  if (idx >= BSZ * S_DIM) return;
  int s = idx & (S_DIM - 1);
  int b = idx >> 10;
  float a = A[s];
  float aC = a;                       // a^CLEN via 7 squarings (CLEN = 128)
#pragma unroll
  for (int q = 0; q < 7; ++q) aC *= aC;
  float hin = h0[s];                  // h_{-1} = h0
  size_t base = ((size_t)b * NCHUNK) * S_DIM + s;
  carries[base] = hin;
  for (int c = 1; c < NCHUNK; ++c) {
    hin = fmaf(aC, hin, localEnd[base + (size_t)(c - 1) * S_DIM]);
    carries[base + (size_t)c * S_DIM] = hin;
  }
}

__global__ void k2c_expand(const uint16_t* __restrict__ uBbf,
                           const float* __restrict__ A,
                           const float* __restrict__ carries,
                           uint16_t* __restrict__ Hbf) {
  int idx = blockIdx.x * blockDim.x + threadIdx.x;
  if (idx >= BSZ * NCHUNK * S_DIM) return;
  int s = idx & (S_DIM - 1);
  int c = (idx >> 10) & (NCHUNK - 1);
  int b = idx >> 14;
  float a = A[s];
  float h = carries[idx];             // state entering this chunk
  size_t base = ((size_t)b * SEQ + (size_t)c * CLEN) * S_DIM + s;
  const uint16_t* p = uBbf + base;
  uint16_t* q = Hbf + base;
#pragma unroll 4
  for (int k = 0; k < CLEN; ++k) {
    h = fmaf(a, h, bf2f(p[(size_t)k * S_DIM]));
    q[(size_t)k * S_DIM] = f2bf(h);
  }
}

// ------------- K3: out = AR(conv w/ M) + shift_kx(H @ C) (WMMA) -------------
// 32x32 tile per wave: 2x2 accumulator block, 2x fragment reuse.
__global__ void k3_out(const uint16_t* __restrict__ Xbf,
                       const uint16_t* __restrict__ Hbf,
                       const uint16_t* __restrict__ Ct,
                       const uint16_t* __restrict__ Mt,
                       float* __restrict__ out) {
  int wave = (int)((blockIdx.x * blockDim.x + threadIdx.x) >> 5);
  int lane = threadIdx.x & 31;
  int tileN = wave & 7;               // 8 col tiles of 32 (OUT_DIM/32)
  int tileM = wave >> 3;              // 1024 row tiles of 32
  if (tileM >= ROWS / 32) return;
  int b  = tileM >> 6;                // 64 row-tiles per batch (SEQ/32)
  int t0 = (tileM & 63) << 5;         // seq offset within batch
  int r = lane & 15, half = lane >> 4;
  int n0 = tileN << 5;

  f32x8 acc[2][2] = {};

  // ---- AR term: sum_{i<KX} M_i @ X shifted back by i (causal zero pad) ----
  const size_t xbase = (size_t)b * SEQ * IN_DIM;
  for (int i = 0; i < KX; ++i) {
    int tr0 = t0 + r - i;                 // upper subtile row (may underflow)
    int tr1 = tr0 + 16;                   // lower subtile row (>= 7, always ok)
    bool ok0 = (tr0 >= 0);
    const uint16_t* xr0 = Xbf + xbase + (size_t)(ok0 ? tr0 : 0) * IN_DIM;
    const uint16_t* xr1 = Xbf + xbase + (size_t)tr1 * IN_DIM;
    const uint16_t* mr0 = Mt + (size_t)i * (OUT_DIM * IN_DIM)
                             + (size_t)(n0 + r) * IN_DIM;
    const uint16_t* mr1 = mr0 + (size_t)16 * IN_DIM;
#pragma unroll 4
    for (int kb = 0; kb < 8; ++kb) {
      Frag16 a0, a1, w0, w1;
      load_frag(a0, xr0, kb * 32, half, ok0);
      load_frag(a1, xr1, kb * 32, half, true);
      load_frag(w0, mr0, kb * 32, half, true);
      load_frag(w1, mr1, kb * 32, half, true);
      acc[0][0] = __builtin_amdgcn_wmma_f32_16x16x32_bf16(false, a0.v, false, w0.v, (short)0, acc[0][0], false, false);
      acc[0][1] = __builtin_amdgcn_wmma_f32_16x16x32_bf16(false, a0.v, false, w1.v, (short)0, acc[0][1], false, false);
      acc[1][0] = __builtin_amdgcn_wmma_f32_16x16x32_bf16(false, a1.v, false, w0.v, (short)0, acc[1][0], false, false);
      acc[1][1] = __builtin_amdgcn_wmma_f32_16x16x32_bf16(false, a1.v, false, w1.v, (short)0, acc[1][1], false, false);
    }
  }

  // ---- shifted LDS term: H[b, t+KX, :] @ C, zero for t+KX >= SEQ ----
  int th0 = t0 + r + KX;
  int th1 = th0 + 16;
  bool ok0 = (th0 < SEQ);
  bool ok1 = (th1 < SEQ);
  const uint16_t* hr0 = Hbf + ((size_t)b * SEQ + (size_t)(ok0 ? th0 : 0)) * S_DIM;
  const uint16_t* hr1 = Hbf + ((size_t)b * SEQ + (size_t)(ok1 ? th1 : 0)) * S_DIM;
  const uint16_t* cr0 = Ct + (size_t)(n0 + r) * S_DIM;
  const uint16_t* cr1 = cr0 + (size_t)16 * S_DIM;
#pragma unroll 4
  for (int kb = 0; kb < 32; ++kb) {
    Frag16 a0, a1, w0, w1;
    load_frag(a0, hr0, kb * 32, half, ok0);
    load_frag(a1, hr1, kb * 32, half, ok1);
    load_frag(w0, cr0, kb * 32, half, true);
    load_frag(w1, cr1, kb * 32, half, true);
    acc[0][0] = __builtin_amdgcn_wmma_f32_16x16x32_bf16(false, a0.v, false, w0.v, (short)0, acc[0][0], false, false);
    acc[0][1] = __builtin_amdgcn_wmma_f32_16x16x32_bf16(false, a0.v, false, w1.v, (short)0, acc[0][1], false, false);
    acc[1][0] = __builtin_amdgcn_wmma_f32_16x16x32_bf16(false, a1.v, false, w0.v, (short)0, acc[1][0], false, false);
    acc[1][1] = __builtin_amdgcn_wmma_f32_16x16x32_bf16(false, a1.v, false, w1.v, (short)0, acc[1][1], false, false);
  }

  float* obase = out + ((size_t)b * SEQ + t0) * OUT_DIM + n0 + r;
#pragma unroll
  for (int mi = 0; mi < 2; ++mi)
#pragma unroll
    for (int ni = 0; ni < 2; ++ni)
#pragma unroll
      for (int i = 0; i < 8; ++i)
        obase[(size_t)(mi * 16 + i + 8 * half) * OUT_DIM + ni * 16]
            = acc[mi][ni][i];
}

// ------------------------------- launcher -----------------------------------
extern "C" void kernel_launch(void* const* d_in, const int* in_sizes, int n_in,
                              void* d_out, int out_size, void* d_ws, size_t ws_size,
                              hipStream_t stream) {
  const float* X  = (const float*)d_in[0];   // [16,2048,256]
  const float* h0 = (const float*)d_in[1];   // [1024]
  const float* A  = (const float*)d_in[2];   // [1024]
  const float* B  = (const float*)d_in[3];   // [256,1024]
  const float* C  = (const float*)d_in[4];   // [1024,256]
  const float* M  = (const float*)d_in[5];   // [256,256,10]
  float* out = (float*)d_out;

  // Workspace layout (bytes), ~149 MiB total, all regions fully rewritten
  // every call (deterministic; safe for graph capture & 0xAA poisoning).
  uint8_t* ws = (uint8_t*)d_ws;
  size_t off = 0;
  uint16_t* Xbf   = (uint16_t*)(ws + off); off += (size_t)ROWS * IN_DIM * 2;          // 16.8 MB
  uint16_t* Btb   = (uint16_t*)(ws + off); off += (size_t)S_DIM * IN_DIM * 2;         // 0.5 MB
  uint16_t* Ctb   = (uint16_t*)(ws + off); off += (size_t)OUT_DIM * S_DIM * 2;        // 0.5 MB
  uint16_t* Mtb   = (uint16_t*)(ws + off); off += (size_t)KX * OUT_DIM * IN_DIM * 2;  // 1.3 MB
  uint16_t* uBbf  = (uint16_t*)(ws + off); off += (size_t)ROWS * S_DIM * 2;           // 67.1 MB
  uint16_t* Hbf   = (uint16_t*)(ws + off); off += (size_t)ROWS * S_DIM * 2;           // 67.1 MB
  float*    localE= (float*)(ws + off);    off += (size_t)BSZ * NCHUNK * S_DIM * 4;   // 1.0 MB
  float*    carry = (float*)(ws + off);    off += (size_t)BSZ * NCHUNK * S_DIM * 4;   // 1.0 MB
  (void)ws_size; (void)in_sizes; (void)n_in; (void)out_size;

  // K0: precision/layout staging
  k0_cvtX <<<(ROWS * IN_DIM + 255) / 256, 256, 0, stream>>>(X, Xbf, ROWS * IN_DIM);
  k0_transB<<<(S_DIM * IN_DIM + 255) / 256, 256, 0, stream>>>(B, Btb);
  k0_transC<<<(OUT_DIM * S_DIM + 255) / 256, 256, 0, stream>>>(C, Ctb);
  k0_transM<<<(KX * OUT_DIM * IN_DIM + 255) / 256, 256, 0, stream>>>(M, Mtb);

  // K1: input projection GEMM (1024 x 32 tiles of 32x32, 1 wave/tile)
  k1_gemm_uB<<<(ROWS / 32) * (S_DIM / 32) / 4, 128, 0, stream>>>(Xbf, Btb, uBbf);

  // K2: chunked parallel scan
  k2a_local <<<(BSZ * NCHUNK * S_DIM + 255) / 256, 256, 0, stream>>>(uBbf, A, localE);
  k2b_scan  <<<(BSZ * S_DIM + 255) / 256, 256, 0, stream>>>(localE, A, h0, carry);
  k2c_expand<<<(BSZ * NCHUNK * S_DIM + 255) / 256, 256, 0, stream>>>(uBbf, A, carry, Hbf);

  // K3: fused AR-conv + shifted C-projection + add (1024 x 8 tiles of 32x32)
  k3_out<<<(ROWS / 32) * (OUT_DIM / 32) / 4, 128, 0, stream>>>(Xbf, Hbf, Ctb, Mtb, out);
}
